// PragnosiaModel_34256659153052
// MI455X (gfx1250) — compile-verified, matrix operand order checked by reading, not run on hardware
//
#include <hip/hip_runtime.h>
#include <hip/hip_bf16.h>

// ---------------------------------------------------------------------------
// Problem sizes (fixed by the reference)
// ---------------------------------------------------------------------------
#define V_SZ   32000
#define H_SZ   1024
#define E_SZ   16
#define K_SEL  2
#define I_SZ   4096
#define R_SZ   512
#define S_SZ   1024
#define B_SZ   2
#define M_TOK  (B_SZ * S_SZ)   // 2048 tokens

// GEMM tiling: 256 threads = 8 wave32, block tile 128x128, k-tile 64.
// Each wave: 64x32 tile = 4x2 grid of 16x16 WMMA accumulators.
#define BM  128
#define BN  128
#define BK  64
#define LDP 72     // padded LDS row stride in bf16 elements (64 + 8)

typedef __bf16 bf16_t;
typedef bf16_t v8bf  __attribute__((ext_vector_type(8)));
typedef bf16_t v16bf __attribute__((ext_vector_type(16)));
typedef float  v8f   __attribute__((ext_vector_type(8)));

// ---------------------------------------------------------------------------
// Workspace layout (bytes)
// ---------------------------------------------------------------------------
#define OFF_H      ((size_t)0)                         // h    f32  [2048,1024]  8 MiB
#define OFF_XN     (OFF_H     + (size_t)M_TOK*H_SZ*4)  // xn   bf16 [2048,1024]  4 MiB
#define OFF_INTER  (OFF_XN    + (size_t)M_TOK*H_SZ*2)  // inter bf16 [2,2048,4096] 32 MiB
#define OFF_COMB   (OFF_INTER + (size_t)K_SEL*M_TOK*I_SZ*2) // comb bf16 [2048,1024] 4 MiB
#define OFF_ACC    (OFF_COMB  + (size_t)M_TOK*H_SZ*2)  // pooled sum f32 [1024]
#define OFF_RIDX   (OFF_ACC   + (size_t)H_SZ*4)        // int[2]
#define OFF_RW     (OFF_RIDX  + (size_t)64)            // float[2]

// ---------------------------------------------------------------------------
// CDNA5 async copy: GLOBAL_LOAD_ASYNC_TO_LDS_B128 (memory -> LDS, no VGPR data,
// tracked by ASYNCcnt).  LDS byte offset = low 32 bits of the generic pointer
// (ISA 10.2).
// ---------------------------------------------------------------------------
__device__ __forceinline__ unsigned lds_off(const void* p) {
  return (unsigned)(uintptr_t)p;
}
__device__ __forceinline__ void async_copy16(unsigned lds_byte, const void* gaddr) {
  asm volatile("global_load_async_to_lds_b128 %0, %1, off"
               :: "v"(lds_byte), "v"(gaddr)
               : "memory");
}
__device__ __forceinline__ void wait_asynccnt0() {
  asm volatile("s_wait_asynccnt 0x0" ::: "memory");
}

// ---------------------------------------------------------------------------
// WMMA fragment helpers
// 16-bit A-matrix 16x32 lane layout (ISA 7.12.2):
//   lanes 0-15 : M=lane,    elems 0-7 = K 0..7,  elems 8-15 = K 16..23
//   lanes 16-31: M=lane-16, elems 0-7 = K 8..15, elems 8-15 = K 24..31
// => two contiguous 16B LDS loads at col offsets khalf*8 and 16+khalf*8.
// B is staged transposed (N-major rows of K), same per-lane pattern with N=row.
// ---------------------------------------------------------------------------
__device__ __forceinline__ v16bf frag_ld(const bf16_t* tile, int row, int cb, int khalf) {
  const bf16_t* p = tile + row * LDP + cb + khalf * 8;
  v8bf lo = *(const v8bf*)(p);
  v8bf hi = *(const v8bf*)(p + 16);
  return __builtin_shufflevector(lo, hi, 0, 1, 2, 3, 4, 5, 6, 7,
                                 8, 9, 10, 11, 12, 13, 14, 15);
}

__device__ __forceinline__ void wave_mma_step(const bf16_t* As, const bf16_t* Bs,
                                              int waveM, int waveN, int lane,
                                              v8f acc[4][2]) {
  const int khalf = lane >> 4;
  const int r     = lane & 15;
#pragma unroll
  for (int ks = 0; ks < 2; ++ks) {          // two 32-wide k substeps of the 64-tile
    const int cb = ks * 32;
    v16bf a[4], b[2];
#pragma unroll
    for (int ms = 0; ms < 4; ++ms)
      a[ms] = frag_ld(As, waveM * 64 + ms * 16 + r, cb, khalf);
#pragma unroll
    for (int ns = 0; ns < 2; ++ns)
      b[ns] = frag_ld(Bs, waveN * 32 + ns * 16 + r, cb, khalf);
#pragma unroll
    for (int ms = 0; ms < 4; ++ms)
#pragma unroll
      for (int ns = 0; ns < 2; ++ns)
        acc[ms][ns] = __builtin_amdgcn_wmma_f32_16x16x32_bf16(
            false, a[ms], false, b[ns], (short)0, acc[ms][ns], false, false);
  }
}

// Async-stage a 128x64 bf16 tile (row-major source) into LDS (padded rows):
// 1024 16-byte chunks, 4 async copies per thread, zero VGPR data movement.
__device__ __forceinline__ void stage_A_async(bf16_t* As, const bf16_t* A, int lda,
                                              int m0, int k0, int tid) {
#pragma unroll
  for (int j = 0; j < 4; ++j) {
    int c   = j * 256 + tid;   // 1024 chunks of 8 bf16 (16 B)
    int row = c >> 3;
    int ch  = c & 7;
    async_copy16(lds_off(As + row * LDP + ch * 8),
                 (const void*)(A + (size_t)(m0 + row) * lda + k0 + ch * 8));
  }
}

// B^T prefetch held in registers (f32 source needs convert+transpose, so it
// must pass through VGPRs): load next tile before compute, store after.
struct BTRegs { float4 f[8]; };

// Source f32 row-major [K x N]; each thread: row kk, 32 consecutive n.
__device__ __forceinline__ void load_BT(BTRegs& r, const float* Bsrc, int ldb,
                                        int k0, int n0, int tid) {
  const int kk = tid >> 2;       // 0..63
  const int q  = tid & 3;
  const float4* s4 = (const float4*)(Bsrc + (size_t)(k0 + kk) * ldb + n0 + q * 32);
#pragma unroll
  for (int i = 0; i < 8; ++i) r.f[i] = s4[i];
}
__device__ __forceinline__ void store_BT(bf16_t* Bs, const BTRegs& r, int tid) {
  const int kk = tid >> 2;
  const int q  = tid & 3;
#pragma unroll
  for (int i = 0; i < 8; ++i) {
    float4 f = r.f[i];
    Bs[(q * 32 + 4 * i + 0) * LDP + kk] = (bf16_t)f.x;
    Bs[(q * 32 + 4 * i + 1) * LDP + kk] = (bf16_t)f.y;
    Bs[(q * 32 + 4 * i + 2) * LDP + kk] = (bf16_t)f.z;
    Bs[(q * 32 + 4 * i + 3) * LDP + kk] = (bf16_t)f.w;
  }
}

// Head GEMM: B^T rows are tok_emb rows (contiguous) -> vectorized both ways.
__device__ __forceinline__ void load_Brow(BTRegs& r, const float* Bsrc, int ldb,
                                          int n0, int k0, int tid) {
  const int nn   = tid >> 1;     // 0..127
  const int half = tid & 1;      // 32 cols each
  const float4* s4 = (const float4*)(Bsrc + (size_t)(n0 + nn) * ldb + k0 + half * 32);
#pragma unroll
  for (int i = 0; i < 8; ++i) r.f[i] = s4[i];
}
__device__ __forceinline__ void store_Brow(bf16_t* Bs, const BTRegs& r, int tid) {
  const int nn   = tid >> 1;
  const int half = tid & 1;
#pragma unroll
  for (int q = 0; q < 4; ++q) {
    float4 f0 = r.f[2 * q], f1 = r.f[2 * q + 1];
    v8bf v = { (bf16_t)f0.x, (bf16_t)f0.y, (bf16_t)f0.z, (bf16_t)f0.w,
               (bf16_t)f1.x, (bf16_t)f1.y, (bf16_t)f1.z, (bf16_t)f1.w };
    *(v8bf*)(Bs + nn * LDP + half * 32 + q * 8) = v;
  }
}

__device__ __forceinline__ float gelu_tanh(float x) {
  float x3 = x * x * x;
  return 0.5f * x * (1.0f + tanhf(0.7978845608f * (x + 0.044715f * x3)));
}

// ---------------------------------------------------------------------------
// Kernel 0: zero the pooled-sum accumulator (1024 f32)
// ---------------------------------------------------------------------------
__global__ void zero_kernel(float* __restrict__ accum) {
  accum[blockIdx.x * 256 + threadIdx.x] = 0.0f;
}

// ---------------------------------------------------------------------------
// Kernel 1: embeddings + pooled sum + LayerNorm.  One block per token.
// ---------------------------------------------------------------------------
__global__ __launch_bounds__(256)
void embed_ln_kernel(const int* __restrict__ ids, const float* __restrict__ tok,
                     const float* __restrict__ pos, const float* __restrict__ g,
                     const float* __restrict__ bb, float* __restrict__ h,
                     bf16_t* __restrict__ xn, float* __restrict__ accum) {
  const int t   = blockIdx.x;       // token index in [0, 2048)
  const int s   = t % S_SZ;
  const int tid = threadIdx.x;      // each thread owns 4 dims
  const int id  = ids[t];

  float4 a = ((const float4*)(tok + (size_t)id * H_SZ))[tid];
  float4 p = ((const float4*)(pos + (size_t)s * H_SZ))[tid];
  float hv[4] = { a.x + p.x, a.y + p.y, a.z + p.z, a.w + p.w };

  float4 hs = { hv[0], hv[1], hv[2], hv[3] };
  ((float4*)(h + (size_t)t * H_SZ))[tid] = hs;

#pragma unroll
  for (int i = 0; i < 4; ++i) atomicAdd(&accum[tid * 4 + i], hv[i]);

  __shared__ float red1[256], red2[256];
  float s1 = hv[0] + hv[1] + hv[2] + hv[3];
  float s2 = hv[0] * hv[0] + hv[1] * hv[1] + hv[2] * hv[2] + hv[3] * hv[3];
  red1[tid] = s1; red2[tid] = s2;
  __syncthreads();
  for (int off = 128; off > 0; off >>= 1) {
    if (tid < off) { red1[tid] += red1[tid + off]; red2[tid] += red2[tid + off]; }
    __syncthreads();
  }
  const float m   = red1[0] * (1.0f / H_SZ);
  const float var = red2[0] * (1.0f / H_SZ) - m * m;
  const float inv = rsqrtf(var + 1e-5f);

  bf16_t* xr = xn + (size_t)t * H_SZ + tid * 4;
#pragma unroll
  for (int i = 0; i < 4; ++i) {
    float v = (hv[i] - m) * inv * g[tid * 4 + i] + bb[tid * 4 + i];
    xr[i] = (bf16_t)v;
  }
}

// ---------------------------------------------------------------------------
// Kernel 2: Hebbian router (tiny).
// ---------------------------------------------------------------------------
__global__ __launch_bounds__(512)
void router_kernel(const float* __restrict__ accum, const float* __restrict__ win,
                   const float* __restrict__ wheb, int* __restrict__ ridx,
                   float* __restrict__ rwts) {
  __shared__ float meanv[H_SZ];
  __shared__ float feat[R_SZ];
  __shared__ float scores[E_SZ];
  const int tid = threadIdx.x;

  meanv[tid]       = accum[tid]       * (1.0f / M_TOK);
  meanv[tid + 512] = accum[tid + 512] * (1.0f / M_TOK);
  __syncthreads();

  float d = 0.0f;
  for (int hh = 0; hh < H_SZ; ++hh) d += meanv[hh] * win[(size_t)hh * R_SZ + tid];
  feat[tid] = tanhf(d);
  __syncthreads();

  if (tid < E_SZ) {
    float sc = 0.0f;
    for (int r = 0; r < R_SZ; ++r) sc += wheb[(size_t)tid * R_SZ + r] * feat[r];
    scores[tid] = sc;
  }
  __syncthreads();

  if (tid == 0) {
    int i0 = 0; float b0 = scores[0];
    for (int e = 1; e < E_SZ; ++e) if (scores[e] > b0) { b0 = scores[e]; i0 = e; }
    int i1 = (i0 == 0) ? 1 : 0; float b1v = scores[i1];
    for (int e = 0; e < E_SZ; ++e)
      if (e != i0 && scores[e] > b1v) { b1v = scores[e]; i1 = e; }
    float z = __expf(b1v - b0);           // b0 >= b1v
    ridx[0] = i0; ridx[1] = i1;
    rwts[0] = 1.0f / (1.0f + z);
    rwts[1] = z / (1.0f + z);
  }
}

// ---------------------------------------------------------------------------
// Kernel 3: GEMM1  inter[kx] = rw[kx] * gelu(xn @ W1[e_kx] + b1[e_kx])   (bf16)
// Double-buffered: async A copies + register-held B prefetch overlap WMMA.
// ---------------------------------------------------------------------------
__global__ __launch_bounds__(256)
void gemm1_kernel(const bf16_t* __restrict__ xn, const float* __restrict__ w1,
                  const float* __restrict__ b1, const int* __restrict__ ridx,
                  const float* __restrict__ rwts, bf16_t* __restrict__ inter) {
  __shared__ __align__(16) bf16_t As[2][BM * LDP];
  __shared__ __align__(16) bf16_t Bs[2][BN * LDP];
  const int tid = threadIdx.x, lane = tid & 31, wave = tid >> 5;
  const int waveM = wave >> 2, waveN = wave & 3;
  const int kx = blockIdx.z;
  const int e  = ridx[kx];
  const float rw = rwts[kx];
  const int n0 = blockIdx.x * BN;
  const int m0 = blockIdx.y * BM;
  const float* W = w1 + (size_t)e * H_SZ * I_SZ;

  v8f acc[4][2] = {};
  BTRegs br;

  // Prologue: stage tile 0 into buffer 0.
  stage_A_async(As[0], xn, H_SZ, m0, 0, tid);
  load_BT(br, W, I_SZ, 0, n0, tid);
  store_BT(Bs[0], br, tid);
  wait_asynccnt0();
  __syncthreads();

  int buf = 0;
  for (int k0 = 0; k0 < H_SZ; k0 += BK) {
    const int nk = k0 + BK;
    if (nk < H_SZ) {
      stage_A_async(As[buf ^ 1], xn, H_SZ, m0, nk, tid);
      load_BT(br, W, I_SZ, nk, n0, tid);
    }
    wave_mma_step(As[buf], Bs[buf], waveM, waveN, lane, acc);
    if (nk < H_SZ) {
      store_BT(Bs[buf ^ 1], br, tid);
      wait_asynccnt0();
    }
    __syncthreads();
    buf ^= 1;
  }

  const float* b1e = b1 + (size_t)e * I_SZ;
  const int cl = lane & 15;
  const int rowBase = m0 + waveM * 64 + (lane >> 4) * 8;
  const int colBase = n0 + waveN * 32;
  bf16_t* dst = inter + (size_t)kx * M_TOK * I_SZ;
#pragma unroll
  for (int ms = 0; ms < 4; ++ms)
#pragma unroll
    for (int ns = 0; ns < 2; ++ns) {
      const int col = colBase + ns * 16 + cl;
      const float bv = b1e[col];
#pragma unroll
      for (int r = 0; r < 8; ++r) {
        const int row = rowBase + ms * 16 + r;
        const float gv = gelu_tanh(acc[ms][ns][r] + bv);
        dst[(size_t)row * I_SZ + col] = (bf16_t)(gv * rw);
      }
    }
}

// ---------------------------------------------------------------------------
// Kernel 4: GEMM2  comb = h + sum_kx inter[kx] @ W2[e_kx] + sum_kx rw*b2[e_kx]
// Flattened 2-expert K loop, double-buffered as above.
// ---------------------------------------------------------------------------
__global__ __launch_bounds__(256)
void gemm2_kernel(const bf16_t* __restrict__ inter, const float* __restrict__ w2,
                  const float* __restrict__ b2, const float* __restrict__ h,
                  const int* __restrict__ ridx, const float* __restrict__ rwts,
                  bf16_t* __restrict__ comb) {
  __shared__ __align__(16) bf16_t As[2][BM * LDP];
  __shared__ __align__(16) bf16_t Bs[2][BN * LDP];
  const int tid = threadIdx.x, lane = tid & 31, wave = tid >> 5;
  const int waveM = wave >> 2, waveN = wave & 3;
  const int n0 = blockIdx.x * BN;
  const int m0 = blockIdx.y * BM;
  const int e0 = ridx[0], e1 = ridx[1];
  const float rw0 = rwts[0], rw1 = rwts[1];
  const float* Wp[K_SEL] = { w2 + (size_t)e0 * I_SZ * H_SZ,
                             w2 + (size_t)e1 * I_SZ * H_SZ };

  const int KPB = I_SZ / BK;          // 64 k-tiles per expert
  const int NIT = K_SEL * KPB;        // 128 total

  v8f acc[4][2] = {};
  BTRegs br;

  stage_A_async(As[0], inter, I_SZ, m0, 0, tid);
  load_BT(br, Wp[0], H_SZ, 0, n0, tid);
  store_BT(Bs[0], br, tid);
  wait_asynccnt0();
  __syncthreads();

  int buf = 0;
  for (int t = 0; t < NIT; ++t) {
    const int nt = t + 1;
    if (nt < NIT) {
      const int kxn = nt / KPB;
      const int k0n = (nt % KPB) * BK;
      stage_A_async(As[buf ^ 1], inter + (size_t)kxn * M_TOK * I_SZ,
                    I_SZ, m0, k0n, tid);
      load_BT(br, Wp[kxn], H_SZ, k0n, n0, tid);
    }
    wave_mma_step(As[buf], Bs[buf], waveM, waveN, lane, acc);
    if (nt < NIT) {
      store_BT(Bs[buf ^ 1], br, tid);
      wait_asynccnt0();
    }
    __syncthreads();
    buf ^= 1;
  }

  const int cl = lane & 15;
  const int rowBase = m0 + waveM * 64 + (lane >> 4) * 8;
  const int colBase = n0 + waveN * 32;
#pragma unroll
  for (int ms = 0; ms < 4; ++ms)
#pragma unroll
    for (int ns = 0; ns < 2; ++ns) {
      const int col = colBase + ns * 16 + cl;
      const float bv = rw0 * b2[(size_t)e0 * H_SZ + col] +
                       rw1 * b2[(size_t)e1 * H_SZ + col];
#pragma unroll
      for (int r = 0; r < 8; ++r) {
        const int row = rowBase + ms * 16 + r;
        const float v = h[(size_t)row * H_SZ + col] + acc[ms][ns][r] + bv;
        comb[(size_t)row * H_SZ + col] = (bf16_t)v;
      }
    }
}

// ---------------------------------------------------------------------------
// Kernel 5: tied head  logits = comb @ tok_emb^T   (f32 out [2048, 32000])
// ---------------------------------------------------------------------------
__global__ __launch_bounds__(256)
void head_kernel(const bf16_t* __restrict__ comb, const float* __restrict__ tok,
                 float* __restrict__ out) {
  __shared__ __align__(16) bf16_t As[2][BM * LDP];
  __shared__ __align__(16) bf16_t Bs[2][BN * LDP];
  const int tid = threadIdx.x, lane = tid & 31, wave = tid >> 5;
  const int waveM = wave >> 2, waveN = wave & 3;
  const int n0 = blockIdx.x * BN;
  const int m0 = blockIdx.y * BM;

  v8f acc[4][2] = {};
  BTRegs br;

  stage_A_async(As[0], comb, H_SZ, m0, 0, tid);
  load_Brow(br, tok, H_SZ, n0, 0, tid);
  store_Brow(Bs[0], br, tid);
  wait_asynccnt0();
  __syncthreads();

  int buf = 0;
  for (int k0 = 0; k0 < H_SZ; k0 += BK) {
    const int nk = k0 + BK;
    if (nk < H_SZ) {
      stage_A_async(As[buf ^ 1], comb, H_SZ, m0, nk, tid);
      load_Brow(br, tok, H_SZ, n0, nk, tid);
    }
    wave_mma_step(As[buf], Bs[buf], waveM, waveN, lane, acc);
    if (nk < H_SZ) {
      store_Brow(Bs[buf ^ 1], br, tid);
      wait_asynccnt0();
    }
    __syncthreads();
    buf ^= 1;
  }

  const int cl = lane & 15;
  const int rowBase = m0 + waveM * 64 + (lane >> 4) * 8;
  const int colBase = n0 + waveN * 32;
#pragma unroll
  for (int ms = 0; ms < 4; ++ms)
#pragma unroll
    for (int ns = 0; ns < 2; ++ns) {
      const int col = colBase + ns * 16 + cl;
#pragma unroll
      for (int r = 0; r < 8; ++r) {
        const int row = rowBase + ms * 16 + r;
        out[(size_t)row * V_SZ + col] = acc[ms][ns][r];
      }
    }
}

// ---------------------------------------------------------------------------
// Launch
// ---------------------------------------------------------------------------
extern "C" void kernel_launch(void* const* d_in, const int* in_sizes, int n_in,
                              void* d_out, int out_size, void* d_ws, size_t ws_size,
                              hipStream_t stream) {
  (void)in_sizes; (void)n_in; (void)out_size; (void)ws_size;
  const int*   ids  = (const int*)  d_in[0];
  const float* tok  = (const float*)d_in[1];
  const float* pos  = (const float*)d_in[2];
  const float* wrin = (const float*)d_in[3];
  const float* wheb = (const float*)d_in[4];
  const float* lng  = (const float*)d_in[5];
  const float* lnb  = (const float*)d_in[6];
  const float* w1   = (const float*)d_in[7];
  const float* b1   = (const float*)d_in[8];
  const float* w2   = (const float*)d_in[9];
  const float* b2   = (const float*)d_in[10];
  float* out = (float*)d_out;

  char* ws = (char*)d_ws;
  float*  hbuf  = (float*) (ws + OFF_H);
  bf16_t* xnbuf = (bf16_t*)(ws + OFF_XN);
  bf16_t* inter = (bf16_t*)(ws + OFF_INTER);
  bf16_t* comb  = (bf16_t*)(ws + OFF_COMB);
  float*  accum = (float*) (ws + OFF_ACC);
  int*    ridx  = (int*)   (ws + OFF_RIDX);
  float*  rwts  = (float*) (ws + OFF_RW);

  zero_kernel<<<dim3(H_SZ / 256), 256, 0, stream>>>(accum);
  embed_ln_kernel<<<dim3(M_TOK), 256, 0, stream>>>(ids, tok, pos, lng, lnb,
                                                   hbuf, xnbuf, accum);
  router_kernel<<<dim3(1), 512, 0, stream>>>(accum, wrin, wheb, ridx, rwts);
  gemm1_kernel<<<dim3(I_SZ / BN, M_TOK / BM, K_SEL), 256, 0, stream>>>(
      xnbuf, w1, b1, ridx, rwts, inter);
  gemm2_kernel<<<dim3(H_SZ / BN, M_TOK / BM), 256, 0, stream>>>(
      inter, w2, b2, hbuf, ridx, rwts, comb);
  head_kernel<<<dim3(V_SZ / BN, M_TOK / BM), 256, 0, stream>>>(comb, tok, out);
}
